// MultiheadAttentionWithKVCache_29738353557941
// MI455X (gfx1250) — matrix operand order repaired and checked by reading, not compiled, not run on hardware
//
#include <hip/hip_runtime.h>
#include <hip/hip_bf16.h>

// ---------------------------------------------------------------------------
// MultiheadAttention with KV cache, CDNA5 (gfx1250) WMMA implementation.
// B=2, LQ=2048, LKV=4096 (2048 cached + 2048 new), E=1024, H=16, D=64.
// Phase 1: QKV projection (f32 -> f16 staging; V staged TRANSPOSED so the
//          attention P*V B-fragments are contiguous), Phase 2: flash
//          attention, Phase 3: output projection (f16 ctx -> f32 out).
// ---------------------------------------------------------------------------

typedef __attribute__((ext_vector_type(16))) _Float16 v16h;
typedef __attribute__((ext_vector_type(8)))  float    v8f;

#define E_DIM   1024
#define NHEAD   16
#define HDIM    64
#define BATCH   2
#define LQ_LEN  2048
#define LKV_LEN 4096
#define LC_LEN  2048

// ---- fragment loaders (contiguous k order, same for A and B, wave32) ---------------
// A-fragment 16x32 f16: lane = row M (lane%16); hf = lane/16.
// elements 0..7  -> K = k0 + hf*16 + i
// elements 8..15 -> K = k0 + hf*16 + 8 + (i-8)
static __device__ __forceinline__ v16h load_a_f32(const float* __restrict__ A,
                                                  int lda, int row, int k0, int hf) {
    const float* p0 = A + (size_t)row * lda + k0 + hf * 16;
    const float* p1 = p0 + 8;
    v16h a;
#pragma unroll
    for (int i = 0; i < 8; ++i) {
        a[i]     = (_Float16)p0[i];
        a[i + 8] = (_Float16)p1[i];
    }
    return a;
}

static __device__ __forceinline__ v16h load_a_f16(const _Float16* __restrict__ A,
                                                  int lda, int row, int k0, int hf) {
    const _Float16* p0 = A + (size_t)row * lda + k0 + hf * 16;
    const _Float16* p1 = p0 + 8;
    v16h a;
#pragma unroll
    for (int i = 0; i < 8; ++i) {
        a[i]     = p0[i];
        a[i + 8] = p1[i];
    }
    return a;
}

// B-fragment 32x16 f16 from row-major weight W[N][K] (B[k][n] = W[n][k]):
// lane holds column n; element i -> K = k0 + (lane/16)*16 + i (contiguous)
static __device__ __forceinline__ v16h load_b_w_f32(const float* __restrict__ W,
                                                    int ldw, int n, int k0) {
    const float* p = W + (size_t)n * ldw + k0;
    v16h b;
#pragma unroll
    for (int i = 0; i < 16; ++i) b[i] = (_Float16)p[i];
    return b;
}

// ---- Phase 1a: GEMM  C_f16[M,1024] = A_f32[M,K] * W^T + bias ---------------
// block = 256 (8 waves, 2M x 4N), block tile 64M x 256N, wave tile 32M x 64N.
__global__ __launch_bounds__(256)
void gemm_bias_f32_to_f16(const float* __restrict__ A, const float* __restrict__ W,
                          const float* __restrict__ bias, _Float16* __restrict__ C,
                          int K) {
    const int lane = threadIdx.x & 31;
    const int wave = threadIdx.x >> 5;
    const int hf   = lane >> 4;
    const int l16  = lane & 15;
    const int m0   = blockIdx.y * 64 + (wave & 1) * 32;
    const int n0   = blockIdx.x * 256 + (wave >> 1) * 64;
    const int ar0  = m0 + l16;
    const int ar1  = m0 + 16 + l16;

    v8f acc[2][4];
#pragma unroll
    for (int r = 0; r < 2; ++r)
#pragma unroll
        for (int j = 0; j < 4; ++j)
#pragma unroll
            for (int v = 0; v < 8; ++v) acc[r][j][v] = 0.0f;

    for (int ks = 0; ks < K; ks += 32) {
        __builtin_prefetch(A + (size_t)ar0 * K + ks + 64, 0, 1);
        __builtin_prefetch(A + (size_t)ar1 * K + ks + 64, 0, 1);
        v16h a0 = load_a_f32(A, K, ar0, ks, hf);
        v16h a1 = load_a_f32(A, K, ar1, ks, hf);
#pragma unroll
        for (int j = 0; j < 4; ++j) {
            v16h b = load_b_w_f32(W, K, n0 + j * 16 + l16, ks + hf * 16);
            acc[0][j] = __builtin_amdgcn_wmma_f32_16x16x32_f16(
                false, a0, false, b, (short)0, acc[0][j], false, false);
            acc[1][j] = __builtin_amdgcn_wmma_f32_16x16x32_f16(
                false, a1, false, b, (short)0, acc[1][j], false, false);
        }
    }

#pragma unroll
    for (int j = 0; j < 4; ++j) {
        const int n = n0 + j * 16 + l16;
        const float bs = bias[n];
#pragma unroll
        for (int r = 0; r < 2; ++r)
#pragma unroll
            for (int v = 0; v < 8; ++v) {
                const int m = m0 + r * 16 + hf * 8 + v;
                C[(size_t)m * E_DIM + n] = (_Float16)(acc[r][j][v] + bs);
            }
    }
}

// ---- Phase 1b: same GEMM but output TRANSPOSED: Ct[n][row_off + m] ---------
// Ct has leading dimension LKV_LEN. Per (r,j) each lane stores 8 contiguous
// f16 along m (C-fragment rows are contiguous per lane) -> 16B stores.
__global__ __launch_bounds__(256)
void gemm_bias_f32_to_f16_t(const float* __restrict__ A, const float* __restrict__ W,
                            const float* __restrict__ bias, _Float16* __restrict__ Ct,
                            int K, int row_off) {
    const int lane = threadIdx.x & 31;
    const int wave = threadIdx.x >> 5;
    const int hf   = lane >> 4;
    const int l16  = lane & 15;
    const int m0   = blockIdx.y * 64 + (wave & 1) * 32;
    const int n0   = blockIdx.x * 256 + (wave >> 1) * 64;
    const int ar0  = m0 + l16;
    const int ar1  = m0 + 16 + l16;

    v8f acc[2][4];
#pragma unroll
    for (int r = 0; r < 2; ++r)
#pragma unroll
        for (int j = 0; j < 4; ++j)
#pragma unroll
            for (int v = 0; v < 8; ++v) acc[r][j][v] = 0.0f;

    for (int ks = 0; ks < K; ks += 32) {
        __builtin_prefetch(A + (size_t)ar0 * K + ks + 64, 0, 1);
        __builtin_prefetch(A + (size_t)ar1 * K + ks + 64, 0, 1);
        v16h a0 = load_a_f32(A, K, ar0, ks, hf);
        v16h a1 = load_a_f32(A, K, ar1, ks, hf);
#pragma unroll
        for (int j = 0; j < 4; ++j) {
            v16h b = load_b_w_f32(W, K, n0 + j * 16 + l16, ks + hf * 16);
            acc[0][j] = __builtin_amdgcn_wmma_f32_16x16x32_f16(
                false, a0, false, b, (short)0, acc[0][j], false, false);
            acc[1][j] = __builtin_amdgcn_wmma_f32_16x16x32_f16(
                false, a1, false, b, (short)0, acc[1][j], false, false);
        }
    }

#pragma unroll
    for (int j = 0; j < 4; ++j) {
        const int n = n0 + j * 16 + l16;
        const float bs = bias[n];
#pragma unroll
        for (int r = 0; r < 2; ++r) {
            _Float16* cp = Ct + (size_t)n * LKV_LEN + row_off + m0 + r * 16 + hf * 8;
#pragma unroll
            for (int v = 0; v < 8; ++v) cp[v] = (_Float16)(acc[r][j][v] + bs);
        }
    }
}

// ---- Phase 3: GEMM  Out_f32[M,1024] = Ctx_f16[M,K] * Wo^T + bo -------------
__global__ __launch_bounds__(256)
void gemm_bias_f16_to_f32(const _Float16* __restrict__ A, const float* __restrict__ W,
                          const float* __restrict__ bias, float* __restrict__ Out,
                          int K) {
    const int lane = threadIdx.x & 31;
    const int wave = threadIdx.x >> 5;
    const int hf   = lane >> 4;
    const int l16  = lane & 15;
    const int m0   = blockIdx.y * 64 + (wave & 1) * 32;
    const int n0   = blockIdx.x * 256 + (wave >> 1) * 64;
    const int ar0  = m0 + l16;
    const int ar1  = m0 + 16 + l16;

    v8f acc[2][4];
#pragma unroll
    for (int r = 0; r < 2; ++r)
#pragma unroll
        for (int j = 0; j < 4; ++j)
#pragma unroll
            for (int v = 0; v < 8; ++v) acc[r][j][v] = 0.0f;

    for (int ks = 0; ks < K; ks += 32) {
        __builtin_prefetch(A + (size_t)ar0 * K + ks + 128, 0, 1);
        __builtin_prefetch(A + (size_t)ar1 * K + ks + 128, 0, 1);
        v16h a0 = load_a_f16(A, K, ar0, ks, hf);
        v16h a1 = load_a_f16(A, K, ar1, ks, hf);
#pragma unroll
        for (int j = 0; j < 4; ++j) {
            v16h b = load_b_w_f32(W, K, n0 + j * 16 + l16, ks + hf * 16);
            acc[0][j] = __builtin_amdgcn_wmma_f32_16x16x32_f16(
                false, a0, false, b, (short)0, acc[0][j], false, false);
            acc[1][j] = __builtin_amdgcn_wmma_f32_16x16x32_f16(
                false, a1, false, b, (short)0, acc[1][j], false, false);
        }
    }

#pragma unroll
    for (int j = 0; j < 4; ++j) {
        const int n = n0 + j * 16 + l16;
        const float bs = bias[n];
#pragma unroll
        for (int r = 0; r < 2; ++r)
#pragma unroll
            for (int v = 0; v < 8; ++v) {
                const int m = m0 + r * 16 + hf * 8 + v;
                Out[(size_t)m * E_DIM + n] = acc[r][j][v] + bs;
            }
    }
}

// ---- Phase 2: flash attention ----------------------------------------------
// grid = (LQ/64, B*H), block = 128 (4 waves). Wave owns 16 q rows, streams
// kv in tiles of 32. S = Q K^T via WMMA, online softmax (f32), P redistributed
// C-layout -> A-layout through per-wave LDS, O += P V via WMMA.
// V is staged transposed (Vt[b][e][kv], ld=LKV) so V B-fragments are
// contiguous 32B per-lane loads, matching the K path.
__global__ __launch_bounds__(128)
void flash_attn(const _Float16* __restrict__ Qh, const _Float16* __restrict__ Kh,
                const _Float16* __restrict__ Vt, _Float16* __restrict__ Ctx) {
    __shared__ _Float16 plds[4][16 * 40];   // 16x32 P tile, padded stride 40

    const int lane = threadIdx.x & 31;
    const int wave = threadIdx.x >> 5;
    const int hf   = lane >> 4;
    const int l16  = lane & 15;

    const int bh = blockIdx.y;
    const int b  = bh / NHEAD;
    const int h  = bh % NHEAD;
    const int q0 = blockIdx.x * 64 + wave * 16;

    const size_t qbase  = (size_t)b * LQ_LEN  * E_DIM + (size_t)h * HDIM;
    const size_t kvbase = (size_t)b * LKV_LEN * E_DIM + (size_t)h * HDIM;
    const float  scale  = 0.125f;  // 1/sqrt(64)

    // Preload Q A-fragments for D=64 (two k-steps of 32), scale folded in.
    v16h qa[2];
    {
        const _Float16* qp = Qh + qbase + (size_t)(q0 + l16) * E_DIM;
#pragma unroll
        for (int ks = 0; ks < 2; ++ks) {
            const _Float16* p0 = qp + ks * 32 + hf * 16;
            const _Float16* p1 = p0 + 8;
            v16h a;
#pragma unroll
            for (int i = 0; i < 8; ++i) {
                a[i]     = (_Float16)((float)p0[i] * scale);
                a[i + 8] = (_Float16)((float)p1[i] * scale);
            }
            qa[ks] = a;
        }
    }

    float mstate[8], lstate[8];
    v8f oacc[4];
#pragma unroll
    for (int v = 0; v < 8; ++v) { mstate[v] = -3.0e38f; lstate[v] = 0.0f; }
#pragma unroll
    for (int j = 0; j < 4; ++j)
#pragma unroll
        for (int v = 0; v < 8; ++v) oacc[j][v] = 0.0f;

    _Float16* P = plds[wave];

    // per-lane base pointers for K (row-major) and Vt (d-major)
    const _Float16* kbase = Kh + kvbase + (size_t)(16 * hf + l16) * E_DIM;
    // Vt row for this lane's d-columns: d = j*16 + l16
    const _Float16* vbase0 = Vt + ((size_t)b * E_DIM + h * HDIM + l16) * LKV_LEN + hf * 16;

    for (int kv0 = 0; kv0 < LKV_LEN; kv0 += 32) {
        // S[16 x 32] = Q * K^T  (B[k][n] = K[kv0+n][d=k], contiguous per lane)
        v8f s[2];
#pragma unroll
        for (int ct = 0; ct < 2; ++ct)
#pragma unroll
            for (int v = 0; v < 8; ++v) s[ct][v] = 0.0f;
#pragma unroll
        for (int ct = 0; ct < 2; ++ct) {
            const _Float16* kp = Kh + kvbase +
                (size_t)(kv0 + ct * 16 + l16) * E_DIM + hf * 16;
            __builtin_prefetch(kp + 32 * E_DIM, 0, 1);
#pragma unroll
            for (int ks = 0; ks < 2; ++ks) {
                v16h bf;
#pragma unroll
                for (int i = 0; i < 16; ++i) bf[i] = kp[ks * 32 + i];
                s[ct] = __builtin_amdgcn_wmma_f32_16x16x32_f16(
                    false, qa[ks], false, bf, (short)0, s[ct], false, false);
            }
        }

        // Online softmax. Row m = hf*8 + v lives in a 16-lane group sharing hf;
        // xor reductions with masks < 16 stay inside that group under wave32.
#pragma unroll
        for (int v = 0; v < 8; ++v) {
            float x = fmaxf(s[0][v], s[1][v]);
#pragma unroll
            for (int msk = 1; msk < 16; msk <<= 1) x = fmaxf(x, __shfl_xor(x, msk, 32));
            const float mn   = fmaxf(mstate[v], x);
            const float corr = __expf(mstate[v] - mn);
            const float p0   = __expf(s[0][v] - mn);
            const float p1   = __expf(s[1][v] - mn);
            float rs = p0 + p1;
#pragma unroll
            for (int msk = 1; msk < 16; msk <<= 1) rs += __shfl_xor(rs, msk, 32);
            lstate[v] = lstate[v] * corr + rs;
            mstate[v] = mn;
#pragma unroll
            for (int j = 0; j < 4; ++j) oacc[j][v] *= corr;
            s[0][v] = p0;
            s[1][v] = p1;
        }

        // C-layout -> row-major LDS (per-wave region; DS ops in-order per wave)
#pragma unroll
        for (int v = 0; v < 8; ++v) {
            const int r = hf * 8 + v;
            P[r * 40 + l16]      = (_Float16)s[0][v];
            P[r * 40 + 16 + l16] = (_Float16)s[1][v];
        }

        // Re-read P as A-fragment (16x32)
        v16h pa;
        {
            const _Float16* p0 = &P[l16 * 40 + hf * 16];
            const _Float16* p1 = p0 + 8;
#pragma unroll
            for (int i = 0; i < 8; ++i) { pa[i] = p0[i]; pa[i + 8] = p1[i]; }
        }

        // O[16 x 64] += P * V.  B[k][n] = Vt[d=n][kv0+k]: contiguous per lane.
#pragma unroll
        for (int j = 0; j < 4; ++j) {
            const _Float16* vp = vbase0 + (size_t)(j * 16) * LKV_LEN + kv0;
            v16h bf;
#pragma unroll
            for (int i = 0; i < 16; ++i) bf[i] = vp[i];
            oacc[j] = __builtin_amdgcn_wmma_f32_16x16x32_f16(
                false, pa, false, bf, (short)0, oacc[j], false, false);
        }
    }
    (void)kbase;

    // Normalize and store ctx (f16)
#pragma unroll
    for (int v = 0; v < 8; ++v) {
        const float inv = 1.0f / lstate[v];
        const int m = q0 + hf * 8 + v;
        _Float16* cp = Ctx + ((size_t)b * LQ_LEN + m) * E_DIM + h * HDIM + l16;
#pragma unroll
        for (int j = 0; j < 4; ++j) cp[j * 16] = (_Float16)(oacc[j][v] * inv);
    }
}

// ---------------------------------------------------------------------------
extern "C" void kernel_launch(void* const* d_in, const int* in_sizes, int n_in,
                              void* d_out, int out_size, void* d_ws, size_t ws_size,
                              hipStream_t stream) {
    (void)in_sizes; (void)n_in; (void)out_size; (void)ws_size;

    const float* query = (const float*)d_in[0];
    const float* key   = (const float*)d_in[1];
    const float* value = (const float*)d_in[2];
    const float* ckey  = (const float*)d_in[3];
    const float* cval  = (const float*)d_in[4];
    const float* wqkv  = (const float*)d_in[5];
    const float* bqkv  = (const float*)d_in[6];
    const float* wo    = (const float*)d_in[7];
    const float* bo    = (const float*)d_in[8];
    float* out = (float*)d_out;

    // workspace carve (f16): Qh 8MiB | Kh 16MiB | Vt 16MiB | Ctx 8MiB
    _Float16* Qh  = (_Float16*)d_ws;
    _Float16* Kh  = Qh + (size_t)BATCH * LQ_LEN  * E_DIM;
    _Float16* Vt  = Kh + (size_t)BATCH * LKV_LEN * E_DIM;   // [B][E][LKV]
    _Float16* Ctx = Vt + (size_t)BATCH * LKV_LEN * E_DIM;

    const float* Wq = wqkv;
    const float* Wk = wqkv + (size_t)E_DIM * E_DIM;
    const float* Wv = wqkv + (size_t)2 * E_DIM * E_DIM;
    const float* bq = bqkv;
    const float* bk = bqkv + E_DIM;
    const float* bv = bqkv + 2 * E_DIM;

    const dim3 blk(256);

    // Q projection over all B*LQ rows at once (contiguous layout)
    {
        dim3 g(E_DIM / 256, (BATCH * LQ_LEN) / 64);
        gemm_bias_f32_to_f16<<<g, blk, 0, stream>>>(query, Wq, bq, Qh, E_DIM);
    }
    // K / V projections: cached rows -> kv[0,2048), new rows -> kv[2048,4096)
    {
        dim3 g(E_DIM / 256, LC_LEN / 64);
        for (int b = 0; b < BATCH; ++b) {
            const size_t inoff = (size_t)b * LC_LEN * E_DIM;
            const size_t kcach = (size_t)b * LKV_LEN * E_DIM;
            const size_t kfrsh = kcach + (size_t)LC_LEN * E_DIM;
            _Float16* vtb = Vt + (size_t)b * E_DIM * LKV_LEN;
            gemm_bias_f32_to_f16<<<g, blk, 0, stream>>>(ckey  + inoff, Wk, bk, Kh + kcach, E_DIM);
            gemm_bias_f32_to_f16<<<g, blk, 0, stream>>>(key   + inoff, Wk, bk, Kh + kfrsh, E_DIM);
            gemm_bias_f32_to_f16_t<<<g, blk, 0, stream>>>(cval  + inoff, Wv, bv, vtb, E_DIM, 0);
            gemm_bias_f32_to_f16_t<<<g, blk, 0, stream>>>(value + inoff, Wv, bv, vtb, E_DIM, LC_LEN);
        }
    }
    // Flash attention
    {
        dim3 g(LQ_LEN / 64, BATCH * NHEAD);
        flash_attn<<<g, dim3(128), 0, stream>>>(Qh, Kh, Vt, Ctx);
    }
    // Output projection
    {
        dim3 g(E_DIM / 256, (BATCH * LQ_LEN) / 64);
        gemm_bias_f16_to_f32<<<g, blk, 0, stream>>>(Ctx, wo, bo, out, E_DIM);
    }
}
